// VQBridge_48361331753023
// MI455X (gfx1250) — matrix operand order repaired and checked by reading, not compiled, not run on hardware
//
#include <hip/hip_runtime.h>

// ---------------------------------------------------------------------------
// CDNA5 (gfx1250) VQ-VAE bridge: every GEMM-shaped stage runs through
// v_wmma_f32_16x16x32_bf16 (wave32).  Fixed shapes: B=16,C=64,H=W=128,D=32,K=512
// One 256-thread block (8 waves) = one full image row (128 pixels); the conv
// A-operand halo (3 rows x CIN, zero-padded) is staged in LDS as bf16 once per
// block -- via GLOBAL_LOAD_ASYNC_TO_LDS_B128 when the input is already bf16 --
// and weights are staged in LDS in K-chunks.
// ---------------------------------------------------------------------------

typedef __attribute__((ext_vector_type(16))) __bf16 v16bf;
typedef __attribute__((ext_vector_type(8)))  float  v8f;
typedef __attribute__((ext_vector_type(8)))  unsigned short us8;
typedef int v4i_t __attribute__((vector_size(4 * sizeof(int))));

#define HH 128
#define WW 128
#define NPIX 262144          // B*H*W
#define DIM 32
#define NCODES 512

#if __has_builtin(__builtin_amdgcn_global_load_async_to_lds_b128)
#define HAVE_ASYNC_LDS 1
#else
#define HAVE_ASYNC_LDS 0
#endif

#define AS1 __attribute__((address_space(1)))
#define AS3 __attribute__((address_space(3)))

__device__ __forceinline__ unsigned short f32_to_bf16(float f) {
    unsigned u = __builtin_bit_cast(unsigned, f);
    unsigned r = u + 0x7FFFu + ((u >> 16) & 1u);   // round-to-nearest-even
    return (unsigned short)(r >> 16);
}
__device__ __forceinline__ float bf16_to_f32(unsigned short h) {
    unsigned u = ((unsigned)h) << 16;
    return __builtin_bit_cast(float, u);
}
__device__ __forceinline__ __bf16 bf_raw(unsigned short h) {
    return __builtin_bit_cast(__bf16, h);
}

// ---------------------------------------------------------------------------
// 3x3 conv as implicit GEMM via WMMA bf16.
//   M = pixels (block = one image row), N = COUT, K = CIN*9 (k = cin*9+dy*3+dx)
// MODE 0: write bf16 NCHW.  MODE 1: write f32 NCHW + bf16 pixel-major.
// MODE 2: write f32 NCHW.   SKIP: fuse a 1x1 conv (32ch) second input.
// ---------------------------------------------------------------------------
template <int CIN, int COUT, int MODE, bool RELU, bool SKIP, bool IN_BF16>
__global__ __launch_bounds__(256) void conv3_wmma(
    const void* __restrict__ in_v,
    const float* __restrict__ w,        // (COUT, CIN, 3, 3)
    const float* __restrict__ bias,     // (COUT)
    const unsigned short* __restrict__ skip_in,  // bf16 NCHW, 32 ch
    const float* __restrict__ skip_w,   // (COUT, 32, 1, 1)
    const float* __restrict__ skip_b,   // (COUT)
    void* __restrict__ out_a,
    unsigned short* __restrict__ out_b)
{
    constexpr int K   = CIN * 9;
    // K-chunk staged in LDS: keep total static LDS under 64 KB per block
    constexpr int KCH = (K <= 288) ? (COUT == 64 ? 96 : 288)
                                   : (COUT == 64 ? 32 : 96);
    constexpr int NCH = K / KCH;
    constexpr int NT  = COUT / 16;      // WMMA n-tiles
    // bf16 inputs use a 16B-aligned halo layout so rows can be DMA'd by B128
    constexpr int XW  = IN_BF16 ? 144 : 130;   // padded row stride (elements)
    constexpr int IB  = IN_BF16 ? 8 : 1;       // column where ix==0 lives

    __shared__ unsigned short Xl[CIN * 3 * XW];       // input halo, bf16
    __shared__ unsigned short Wl[COUT * KCH];         // weight K-chunk, bf16
    __shared__ unsigned short SWl[SKIP ? COUT * 32 : 2];

    const int tid  = threadIdx.x;
    const int wave = tid >> 5;
    const int lane = tid & 31;
    const int lm   = lane & 15;
    const bool hi  = lane >= 16;
    const int koffA = hi ? 8 : 0;       // A-frag K offset (ISA 16-bit A layout)
    const int koffB = hi ? 8 : 0;       // B-frag K offset

    const float*          inf = IN_BF16 ? nullptr : (const float*)in_v;
    const unsigned short* inb = IN_BF16 ? (const unsigned short*)in_v : nullptr;

    // block = one image row
    const int b  = blockIdx.x >> 7;
    const int y  = blockIdx.x & 127;
    const int xA = wave * 16 + lm;      // A-gather pixel x for this lane

    // warm caches for the halo rows (global_prefetch_b8)
    {
        int iy0 = y > 0 ? y - 1 : 0;
        for (int c = tid; c < CIN; c += 256) {
            size_t gi = (((size_t)b * CIN + c) * HH + iy0) * WW;
            __builtin_prefetch((const char*)in_v + gi * (IN_BF16 ? 2 : 4), 0, 1);
        }
    }

    if (SKIP) {
        for (int i = tid; i < COUT * 32; i += 256)
            SWl[i] = f32_to_bf16(skip_w[i]);
    }

    // ---- stage input halo: Xl[(cin*3+dy)*XW + IB + ix], zero-padded ----
#if HAVE_ASYNC_LDS
    if constexpr (IN_BF16) {
        // interior: async DMA, 16B chunks (ASYNCcnt); OOB rows zero-filled
        for (int i = tid; i < CIN * 3 * 16; i += 256) {
            int row = i >> 4;                 // cin*3 + dy
            int ch  = i & 15;                 // 16 chunks of 8 bf16 per row
            int cin = row / 3;
            int dy  = row - cin * 3;
            int iy  = y + dy - 1;
            unsigned short* ldst = &Xl[row * XW + IB + ch * 8];
            if (iy >= 0 && iy < HH) {
                const unsigned short* gsrc =
                    inb + (((size_t)b * CIN + cin) * HH + iy) * WW + ch * 8;
                __builtin_amdgcn_global_load_async_to_lds_b128(
                    (AS1 v4i_t*)(unsigned long long)(const void*)gsrc,
                    (AS3 v4i_t*)(unsigned)(unsigned long long)(void*)ldst,
                    0, 0);
            } else {
                *(us8*)ldst = (us8)(unsigned short)0;
            }
        }
        // zero the pad columns (ix=-1 at col IB-1, ix=128 at col IB+128, edges)
        for (int i = tid; i < CIN * 3 * 16; i += 256) {
            int row = i >> 4;
            int c   = i & 15;
            int col = (c < IB) ? c : (128 + c);   // 0..7 and 136..143
            Xl[row * XW + col] = 0;
        }
        asm volatile("s_wait_asynccnt 0" ::: "memory");
    } else
#endif
    {
        for (int i = tid; i < CIN * 3 * XW; i += 256) {
            int cin = i / (3 * XW);
            int rem = i - cin * 3 * XW;
            int dy  = rem / XW;
            int xx  = rem - dy * XW;
            int iy  = y + dy - 1;
            int ix  = xx - IB;
            unsigned short hv = 0;
            if (iy >= 0 && iy < HH && ix >= 0 && ix < WW) {
                int gi = ((b * CIN + cin) * HH + iy) * WW + ix;
                hv = IN_BF16 ? inb[gi] : f32_to_bf16(inf[gi]);
            }
            Xl[i] = hv;
        }
    }

    v8f acc[NT];
#pragma unroll
    for (int t = 0; t < NT; ++t)
        acc[t] = (v8f){0.f, 0.f, 0.f, 0.f, 0.f, 0.f, 0.f, 0.f};

    for (int kc = 0; kc < NCH; ++kc) {
        __syncthreads();
        // stage one K-chunk of weights: Wl[cout*KCH + kk] = w[cout*K + kc*KCH + kk]
        for (int i = tid; i < COUT * KCH; i += 256) {
            int cout = i / KCH;
            int kk   = i - cout * KCH;
            Wl[i] = f32_to_bf16(w[cout * K + kc * KCH + kk]);
        }
        if (kc + 1 < NCH && tid < COUT)
            __builtin_prefetch(&w[tid * K + (kc + 1) * KCH], 0, 1);
        __syncthreads();

        for (int ks = 0; ks < KCH / 32; ++ks) {
            const int kbase = kc * KCH + ks * 32;
            // ---- A fragment: bounds-free gather from LDS halo ----
            v16bf afr;
#pragma unroll
            for (int j = 0; j < 16; ++j) {
                int k   = kbase + koffA + (j < 8 ? j : j + 8);
                int cin = k / 9;
                int r9  = k - cin * 9;
                int dy  = r9 / 3;
                int dx  = r9 - dy * 3;
                afr[j] = bf_raw(Xl[(cin * 3 + dy) * XW + IB + xA + dx - 1]);
            }
            // ---- B fragments from LDS + WMMA per n-tile ----
#pragma unroll
            for (int t = 0; t < NT; ++t) {
                int bi = (t * 16 + lm) * KCH + ks * 32 + koffB;
                us8 c0 = *(const us8*)&Wl[bi];
                us8 c1 = *(const us8*)&Wl[bi + 16];
                v16bf bfr;
#pragma unroll
                for (int j = 0; j < 8; ++j) {
                    bfr[j]     = bf_raw(c0[j]);
                    bfr[j + 8] = bf_raw(c1[j]);
                }
                acc[t] = __builtin_amdgcn_wmma_f32_16x16x32_bf16(
                    false, afr, false, bfr, (short)0, acc[t], false, false);
            }
        }
    }

    if (SKIP) {
        // fused 1x1 conv over 32-ch second input (one K=32 WMMA per n-tile)
        v16bf afr;
#pragma unroll
        for (int j = 0; j < 16; ++j) {
            int cin = koffA + (j < 8 ? j : j + 8);
            unsigned short hv = skip_in[((b * 32 + cin) * HH + y) * WW + xA];
            afr[j] = bf_raw(hv);
        }
#pragma unroll
        for (int t = 0; t < NT; ++t) {
            int bi = (t * 16 + lm) * 32 + koffB;
            us8 c0 = *(const us8*)&SWl[bi];
            us8 c1 = *(const us8*)&SWl[bi + 16];
            v16bf bfr;
#pragma unroll
            for (int j = 0; j < 8; ++j) {
                bfr[j]     = bf_raw(c0[j]);
                bfr[j + 8] = bf_raw(c1[j]);
            }
            acc[t] = __builtin_amdgcn_wmma_f32_16x16x32_bf16(
                false, afr, false, bfr, (short)0, acc[t], false, false);
        }
    }

    // ---- epilogue: C/D layout -> lane l, VGPR r: M = r + (hi?8:0), N = lm ----
#pragma unroll
    for (int t = 0; t < NT; ++t) {
        int n = t * 16 + lm;
        float bv = bias[n] + (SKIP ? skip_b[n] : 0.f);
#pragma unroll
        for (int r = 0; r < 8; ++r) {
            int x = wave * 16 + r + (hi ? 8 : 0);
            float v = acc[t][r] + bv;
            if (RELU) v = fmaxf(v, 0.f);
            int oi = ((b * COUT + n) * HH + y) * WW + x;
            if (MODE == 0) {
                ((unsigned short*)out_a)[oi] = f32_to_bf16(v);
            } else if (MODE == 1) {
                ((float*)out_a)[oi] = v;
                int P = blockIdx.x * 128 + x;
                out_b[P * COUT + n] = f32_to_bf16(v);   // pixel-major bf16
            } else {
                ((float*)out_a)[oi] = v;
            }
        }
    }
}

// ---------------------------------------------------------------------------
// VQ stage: WMMA GEMM z_flat(262144x32) x codebook^T(32x512), argmin of
// -2 z.c + ||c||^2 (z^2 term constant per row), shfl_xor reduction (width 16
// matches the WMMA C layout), then z_q / indices / EMA-statistics scatter.
// ---------------------------------------------------------------------------
__global__ __launch_bounds__(256) void vq_wmma(
    const unsigned short* __restrict__ zf,   // bf16 pixel-major (NPIX, 32)
    const float* __restrict__ codebook,      // (512, 32) f32
    const float* __restrict__ ze,            // f32 NCHW (B,32,H,W)
    float* __restrict__ zq_out,              // d_out sec 1: f32 NCHW
    float* __restrict__ idx_out,             // d_out sec 4: float indices
    unsigned short* __restrict__ zq_bf,      // bf16 NCHW for decoder
    float* __restrict__ counts,              // (512) ws, pre-zeroed
    float* __restrict__ dw,                  // (512,32) ws, pre-zeroed
    float* __restrict__ commit)              // scalar ws, pre-zeroed
{
    __shared__ unsigned short CB[NCODES * DIM];  // bf16 codebook (32 KB)
    __shared__ float C2[NCODES];

    const int tid = threadIdx.x;
    for (int i = tid; i < NCODES * DIM; i += 256)
        CB[i] = f32_to_bf16(codebook[i]);
    for (int c = tid; c < NCODES; c += 256) {
        float s = 0.f;
        for (int d = 0; d < DIM; ++d) {
            float v = codebook[c * DIM + d];
            s += v * v;
        }
        C2[c] = s;
    }
    __syncthreads();

    const int wave = tid >> 5;
    const int lane = tid & 31;
    const int lm   = lane & 15;
    const bool hi  = lane >= 16;
    const int koffA = hi ? 8 : 0;
    const int koffB = hi ? 8 : 0;
    const int pbase = blockIdx.x * 128 + wave * 16;

    // A fragment: this lane's pixel row, two 16B vector loads
    const unsigned short* row = zf + (size_t)(pbase + lm) * DIM;
    us8 a0 = *(const us8*)(row + koffA);
    us8 a1 = *(const us8*)(row + 16 + koffA);
    v16bf afr;
#pragma unroll
    for (int j = 0; j < 8; ++j) {
        afr[j]     = bf_raw(a0[j]);
        afr[j + 8] = bf_raw(a1[j]);
    }

    float best[8];
    int   bidx[8];
#pragma unroll
    for (int r = 0; r < 8; ++r) { best[r] = 3.4e38f; bidx[r] = 0; }

    for (int n0 = 0; n0 < NCODES; n0 += 16) {
        int bi = (n0 + lm) * DIM + koffB;
        us8 c0 = *(const us8*)&CB[bi];
        us8 c1 = *(const us8*)&CB[bi + 16];
        v16bf bfr;
#pragma unroll
        for (int j = 0; j < 8; ++j) {
            bfr[j]     = bf_raw(c0[j]);
            bfr[j + 8] = bf_raw(c1[j]);
        }
        v8f acc = (v8f){0.f, 0.f, 0.f, 0.f, 0.f, 0.f, 0.f, 0.f};
        acc = __builtin_amdgcn_wmma_f32_16x16x32_bf16(
            false, afr, false, bfr, (short)0, acc, false, false);
        float c2v = C2[n0 + lm];
        int   n   = n0 + lm;
#pragma unroll
        for (int r = 0; r < 8; ++r) {
            float d = c2v - 2.f * acc[r];
            if (d < best[r] || (d == best[r] && n < bidx[r])) {
                best[r] = d; bidx[r] = n;
            }
        }
    }

    // reduce over the 16 lanes sharing each pixel (C-layout half-waves)
#pragma unroll
    for (int r = 0; r < 8; ++r) {
#pragma unroll
        for (int off = 1; off < 16; off <<= 1) {
            float ov = __shfl_xor(best[r], off, 16);
            int   oi = __shfl_xor(bidx[r], off, 16);
            if (ov < best[r] || (ov == best[r] && oi < bidx[r])) {
                best[r] = ov; bidx[r] = oi;
            }
        }
    }

    if (lm == 0) {
#pragma unroll 1
        for (int r = 0; r < 8; ++r) {
            int P   = pbase + r + (hi ? 8 : 0);
            int idx = bidx[r];
            idx_out[P] = (float)idx;
            atomicAdd(&counts[idx], 1.0f);
            int b = P >> 14, y = (P >> 7) & 127, x = P & 127;
            float cl = 0.f;
            for (int d = 0; d < DIM; ++d) {
                float cv  = codebook[idx * DIM + d];
                int   oi  = ((b * DIM + d) * HH + y) * WW + x;
                float zev = ze[oi];
                zq_out[oi] = cv;                   // straight-through value
                zq_bf[oi]  = f32_to_bf16(cv);
                atomicAdd(&dw[idx * DIM + d], zev);
                float df = zev - cv;
                cl += df * df;
            }
            atomicAdd(commit, cl);
        }
    }
}

__global__ __launch_bounds__(256) void init_zero(float* p, int n) {
    int i = blockIdx.x * 256 + threadIdx.x;
    if (i < n) p[i] = 0.f;
}

__global__ __launch_bounds__(256) void ema_finalize(
    const float* __restrict__ counts, const float* __restrict__ dw,
    const float* __restrict__ commit,
    const float* __restrict__ ema_count, const float* __restrict__ ema_weight,
    float* __restrict__ out_commit, float* __restrict__ out_cb,
    float* __restrict__ out_cnt, float* __restrict__ out_w)
{
    int i = blockIdx.x * 256 + threadIdx.x;
    if (i < NCODES * DIM) {
        int k = i >> 5;
        float nc = 0.99f * ema_count[k] + 0.01f * counts[k];
        float nw = 0.99f * ema_weight[i] + 0.01f * dw[i];
        out_w[i]  = nw;
        out_cb[i] = nw / fmaxf(nc, 1.0f);
        if (i < NCODES) out_cnt[i] = 0.99f * ema_count[i] + 0.01f * counts[i];
        if (i == 0) out_commit[0] = commit[0] / 8388608.0f;
    }
}

// ---------------------------------------------------------------------------
extern "C" void kernel_launch(void* const* d_in, const int* in_sizes, int n_in,
                              void* d_out, int out_size, void* d_ws, size_t ws_size,
                              hipStream_t stream) {
    (void)in_sizes; (void)n_in; (void)out_size; (void)ws_size;

    const float* h         = (const float*)d_in[0];
    const float* q1_w      = (const float*)d_in[1];
    const float* q1_b      = (const float*)d_in[2];
    const float* q2_w      = (const float*)d_in[3];
    const float* q2_b      = (const float*)d_in[4];
    const float* r1_w      = (const float*)d_in[5];
    const float* r1_b      = (const float*)d_in[6];
    const float* r2_w      = (const float*)d_in[7];
    const float* r2_b      = (const float*)d_in[8];
    const float* skip_w    = (const float*)d_in[9];
    const float* skip_b    = (const float*)d_in[10];
    const float* codebook  = (const float*)d_in[11];
    const float* ema_count = (const float*)d_in[12];
    const float* ema_wt    = (const float*)d_in[13];

    float* out = (float*)d_out;
    // output sections (flat, reference return order)
    float* o_zq     = out;                         // 8388608
    float* o_hhat   = out + 8388608;               // 16777216
    float* o_commit = out + 25165824;              // 1
    float* o_idx    = out + 25165825;              // 262144
    float* o_cb     = out + 25427969;              // 16384
    float* o_cnt    = out + 25444353;              // 512
    float* o_w      = out + 25444865;              // 16384

    // workspace layout
    char* ws = (char*)d_ws;
    float* cnt    = (float*)ws;                    // 512
    float* dw     = cnt + 512;                     // 16384
    float* commit = dw + 16384;                    // 1
    size_t off = 67840;                            // 256B-aligned past accumulators
    unsigned short* z_bf   = (unsigned short*)(ws + off); off += 16777216; // z (bf16 NCHW)
    float*          z_e    = (float*)(ws + off);          off += 33554432; // z_e f32 NCHW
    unsigned short* z_flat = (unsigned short*)(ws + off); off += 16777216; // z_e bf16 pixel-major
    unsigned short* zq_bf  = (unsigned short*)(ws + off); off += 16777216; // z_q bf16 NCHW
    unsigned short* r_bf   = (unsigned short*)(ws + off); off += 33554432; // r bf16 NCHW

    const int blocks = NPIX / 128;   // 2048: one block per image row

    init_zero<<<67, 256, 0, stream>>>(cnt, 512 + 16384 + 1);

    // encoder
    conv3_wmma<64, 32, 0, true,  false, false><<<blocks, 256, 0, stream>>>(
        h, q1_w, q1_b, nullptr, nullptr, nullptr, z_bf, nullptr);
    conv3_wmma<32, 32, 1, false, false, true ><<<blocks, 256, 0, stream>>>(
        z_bf, q2_w, q2_b, nullptr, nullptr, nullptr, z_e, z_flat);

    // vector quantization + EMA statistics
    vq_wmma<<<blocks, 256, 0, stream>>>(z_flat, codebook, z_e, o_zq, o_idx,
                                        zq_bf, cnt, dw, commit);

    // decoder (skip 1x1 fused into r2)
    conv3_wmma<32, 64, 0, true,  false, true ><<<blocks, 256, 0, stream>>>(
        zq_bf, r1_w, r1_b, nullptr, nullptr, nullptr, r_bf, nullptr);
    conv3_wmma<64, 64, 2, false, true,  true ><<<blocks, 256, 0, stream>>>(
        r_bf, r2_w, r2_b, zq_bf, skip_w, skip_b, o_hhat, nullptr);

    // EMA codebook update + commit-loss mean
    ema_finalize<<<64, 256, 0, stream>>>(cnt, dw, commit, ema_count, ema_wt,
                                         o_commit, o_cb, o_cnt, o_w);
}
